// Perceiver_29540785061956
// MI455X (gfx1250) — compile-verified, matrix operand order checked by reading, not run on hardware
//
#include <hip/hip_runtime.h>
#include <hip/hip_bf16.h>
#include <math.h>

typedef __attribute__((ext_vector_type(16))) __bf16 v16bf;
typedef __attribute__((ext_vector_type(8)))  float  v8f;

#define DD    4096
#define NN    1792
#define BTC   16        // B*T
#define HH    8
#define DHD   96
#define INNERC 768
#define KVC   1536
#define FFD   8192
#define ROWSC (BTC*NN)  // 28672
#define LROWS 256       // B*T*NL
#define QSCALE 0.1020620726159657f  // 96^-0.5

// ---------------------------------------------------------------------------
// Row LayerNorm over D=4096: one workgroup per row, 256 threads x 16 elems.
// outf != nullptr -> fp32 output, else bf16 output.
// ---------------------------------------------------------------------------
__global__ __launch_bounds__(256) void ln_rows_k(
    const float* __restrict__ in, const float* __restrict__ gam,
    const float* __restrict__ bet, float* __restrict__ outf,
    __bf16* __restrict__ outb) {
  __shared__ float s_sum[256];
  __shared__ float s_sq[256];
  const int row = blockIdx.x, t = threadIdx.x;
  const float* p = in + (size_t)row * DD;
  float x[16], sum = 0.f, sq = 0.f;
#pragma unroll
  for (int u = 0; u < 16; ++u) {
    x[u] = p[t + 256 * u];
    sum += x[u];
    sq  += x[u] * x[u];
  }
  s_sum[t] = sum; s_sq[t] = sq;
  __syncthreads();
  for (int off = 128; off > 0; off >>= 1) {
    if (t < off) { s_sum[t] += s_sum[t + off]; s_sq[t] += s_sq[t + off]; }
    __syncthreads();
  }
  const float mean = s_sum[0] * (1.0f / DD);
  const float var  = s_sq[0] * (1.0f / DD) - mean * mean;
  const float rstd = rsqrtf(var + 1e-5f);
#pragma unroll
  for (int u = 0; u < 16; ++u) {
    const int c = t + 256 * u;
    const float y = (x[u] - mean) * rstd * gam[c] + bet[c];
    if (outf) outf[(size_t)row * DD + c] = y;
    else      outb[(size_t)row * DD + c] = (__bf16)y;
  }
}

// ---------------------------------------------------------------------------
// wkv (fp32, [4096][1536] K-major) -> wkvT (bf16, [1536][4096] N-major)
// ---------------------------------------------------------------------------
__global__ __launch_bounds__(256) void cvt_wkvT_k(
    const float* __restrict__ wkv, __bf16* __restrict__ wkvT) {
  const size_t idx = (size_t)blockIdx.x * 256 + threadIdx.x;  // n*4096 + k
  const int k = (int)(idx & (DD - 1));
  const int n = (int)(idx >> 12);
  wkvT[(size_t)n * DD + k] = (__bf16)wkv[(size_t)k * KVC + n];
}

// ---------------------------------------------------------------------------
// Fragment loader: 16-bit A-style layout from LDS tile with row stride 32.
// Lane l: row = base + (l&15); element e -> K = e + 8*(e>=8) + 8*(l>=16).
// => two aligned 16B ds reads at k-offsets 8*hi and 16+8*hi.
// ---------------------------------------------------------------------------
union Frag16 { v16bf v; uint4 u[2]; };

__device__ __forceinline__ void load_frag(Frag16& f, const __bf16* lds,
                                          int row, int hi) {
  f.u[0] = *reinterpret_cast<const uint4*>(&lds[row * 32 + 8 * hi]);
  f.u[1] = *reinterpret_cast<const uint4*>(&lds[row * 32 + 16 + 8 * hi]);
}

// LDS byte address of a shared-memory pointer: flat LDS addresses carry the
// LDS offset in the low 32 bits (aperture in the high bits), so truncate.
__device__ __forceinline__ unsigned lds_addr_of(const void* p) {
  return (unsigned)(uintptr_t)p;
}

// Async 16B global->LDS copy (GVS mode: saddr base + 32-bit lane offset).
// Tracked by ASYNCcnt; completion enforced with s_wait_asynccnt.
__device__ __forceinline__ void async_b128(unsigned lds_byte, unsigned gbl_byte,
                                           const void* base) {
  asm volatile("global_load_async_to_lds_b128 %0, %1, %2"
               :: "v"(lds_byte), "v"(gbl_byte), "s"(base)
               : "memory");
}

// ---------------------------------------------------------------------------
// kv = xm_bf16 (28672x4096) @ wkvT_bf16 -> kv_bf16 (28672x1536)
// Tile M=128, N=64, K=32. 8 waves in 4x2 grid, each wave 32x32 (2x2 WMMA).
// Double-buffered async global->LDS staging: tile k+1 copies in flight while
// tile k WMMAs run; wave waits s_wait_asynccnt<=3 (its 3 newest issues).
// ---------------------------------------------------------------------------
__global__ __launch_bounds__(256) void gemm_kv_k(
    const __bf16* __restrict__ A, const __bf16* __restrict__ Bt,
    __bf16* __restrict__ C) {
  __shared__ __bf16 As[2][128 * 32];
  __shared__ __bf16 Bs[2][64 * 32];
  const int t = threadIdx.x;
  const int wave = t >> 5, lane = t & 31;
  const int hi = lane >> 4, ln16 = lane & 15;
  const int wr = wave >> 1, wc = wave & 1;
  const int m0 = blockIdx.y * 128;
  const int n0 = blockIdx.x * 64;

  const unsigned ldsA = lds_addr_of(&As[0][0]);
  const unsigned ldsB = lds_addr_of(&Bs[0][0]);
  const unsigned ABYTES = 128 * 32 * 2;
  const unsigned BBYTES = 64 * 32 * 2;

  // per-thread staging roles (16B chunks)
  const int rA0 = t >> 2, pA0 = t & 3;       // A chunk t
  const int rA1 = rA0 + 64;                  // A chunk t+256
  const int rB  = t >> 2, pB = t & 3;        // B chunk t

  const unsigned gA0 = (unsigned)(m0 + rA0) * (DD * 2u) + (unsigned)pA0 * 16u;
  const unsigned gA1 = (unsigned)(m0 + rA1) * (DD * 2u) + (unsigned)pA0 * 16u;
  const unsigned gB  = (unsigned)(n0 + rB)  * (DD * 2u) + (unsigned)pB  * 16u;
  const unsigned lA0 = (unsigned)(rA0 * 64 + pA0 * 16);
  const unsigned lA1 = (unsigned)(rA1 * 64 + pA0 * 16);
  const unsigned lB  = (unsigned)(rB * 64 + pB * 16);

  v8f acc[2][2] = {};

  // prologue: tile 0 -> buffer 0
  {
    const unsigned kb0 = 0;
    async_b128(ldsA + lA0, gA0 + kb0, A);
    async_b128(ldsA + lA1, gA1 + kb0, A);
    async_b128(ldsB + lB,  gB  + kb0, Bt);
  }

  const int NK = DD / 32;  // 128
  for (int kb = 0; kb < NK; ++kb) {
    const int cur = kb & 1;
    if (kb + 1 < NK) {
      const unsigned ko = (unsigned)(kb + 1) * 64u;  // 32 bf16 = 64 bytes
      const unsigned abuf = (unsigned)(cur ^ 1) * ABYTES;
      const unsigned bbuf = (unsigned)(cur ^ 1) * BBYTES;
      async_b128(ldsA + abuf + lA0, gA0 + ko, A);
      async_b128(ldsA + abuf + lA1, gA1 + ko, A);
      async_b128(ldsB + bbuf + lB,  gB  + ko, Bt);
      asm volatile("s_wait_asynccnt 0x3" ::: "memory");  // tile kb complete
    } else {
      asm volatile("s_wait_asynccnt 0x0" ::: "memory");
    }
    __syncthreads();  // all waves see tile kb in LDS

    const __bf16* Ab = &As[cur][0];
    const __bf16* Bb = &Bs[cur][0];
    Frag16 a[2], b[2];
#pragma unroll
    for (int i = 0; i < 2; ++i) load_frag(a[i], Ab, wr * 32 + i * 16 + ln16, hi);
#pragma unroll
    for (int j = 0; j < 2; ++j) load_frag(b[j], Bb, wc * 32 + j * 16 + ln16, hi);
#pragma unroll
    for (int i = 0; i < 2; ++i)
#pragma unroll
      for (int j = 0; j < 2; ++j)
        acc[i][j] = __builtin_amdgcn_wmma_f32_16x16x32_bf16(
            false, a[i].v, false, b[j].v, (short)0, acc[i][j], false, false);

    __syncthreads();  // all reads of buf cur done before it is refilled
  }

#pragma unroll
  for (int i = 0; i < 2; ++i)
#pragma unroll
    for (int j = 0; j < 2; ++j)
#pragma unroll
      for (int v = 0; v < 8; ++v) {
        const int row = m0 + wr * 32 + i * 16 + v + 8 * hi;
        const int col = n0 + wc * 32 + j * 16 + ln16;
        C[(size_t)row * KVC + col] = (__bf16)acc[i][j][v];
      }
}

// ---------------------------------------------------------------------------
// Generic 256-row GEMM: out = act(A_bf16[256][K] @ B_fp32[K][Nf]) + residual.
// A tile staged with async global->LDS (overlaps the fp32->bf16 B conversion,
// which must ride VALU). Each weight element is read from HBM exactly once.
// resmode: 0 none, 1 res[(row&15)*Nf+col] (broadcast latents), 2 res[row*Nf+col]
// act: 1 -> exact GELU. outf fp32 or outb bf16.
// ---------------------------------------------------------------------------
__global__ __launch_bounds__(256) void gemm_wb_k(
    const __bf16* __restrict__ A, int K, const float* __restrict__ B, int Nf,
    const float* __restrict__ res, int resmode, int act,
    float* __restrict__ outf, __bf16* __restrict__ outb) {
  __shared__ __bf16 As[256 * 32];
  __shared__ __bf16 Bs[64 * 32];
  const int t = threadIdx.x;
  const int wave = t >> 5, lane = t & 31;
  const int hi = lane >> 4, ln16 = lane & 15;
  const int n0 = blockIdx.x * 64;

  const unsigned ldsA = lds_addr_of(&As[0]);
  v8f acc[2][4] = {};

  for (int k0 = 0; k0 < K; k0 += 32) {
    if (k0 + 32 < K)
      __builtin_prefetch(B + (size_t)(k0 + 32 + (t >> 3)) * Nf + n0, 0, 1);
    __syncthreads();  // previous tile fully consumed
#pragma unroll
    for (int s = 0; s < 4; ++s) {
      const int c = t + 256 * s;
      const int row = c >> 2, part = c & 3;
      const unsigned goff = (unsigned)row * ((unsigned)K * 2u) +
                            (unsigned)(k0 + part * 8) * 2u;
      const unsigned loff = ldsA + (unsigned)(row * 64 + part * 16);
      async_b128(loff, goff, A);
    }
    {
      const int k  = t >> 3;
      const int ng = (t & 7) * 8;
      const float* src = B + (size_t)(k0 + k) * Nf + n0 + ng;
      const float4 f0 = *reinterpret_cast<const float4*>(src);
      const float4 f1 = *reinterpret_cast<const float4*>(src + 4);
      Bs[(ng + 0) * 32 + k] = (__bf16)f0.x;
      Bs[(ng + 1) * 32 + k] = (__bf16)f0.y;
      Bs[(ng + 2) * 32 + k] = (__bf16)f0.z;
      Bs[(ng + 3) * 32 + k] = (__bf16)f0.w;
      Bs[(ng + 4) * 32 + k] = (__bf16)f1.x;
      Bs[(ng + 5) * 32 + k] = (__bf16)f1.y;
      Bs[(ng + 6) * 32 + k] = (__bf16)f1.z;
      Bs[(ng + 7) * 32 + k] = (__bf16)f1.w;
    }
    asm volatile("s_wait_asynccnt 0x0" ::: "memory");
    __syncthreads();
    Frag16 a[2], b[4];
#pragma unroll
    for (int i = 0; i < 2; ++i) load_frag(a[i], As, wave * 32 + i * 16 + ln16, hi);
#pragma unroll
    for (int j = 0; j < 4; ++j) load_frag(b[j], Bs, j * 16 + ln16, hi);
#pragma unroll
    for (int i = 0; i < 2; ++i)
#pragma unroll
      for (int j = 0; j < 4; ++j)
        acc[i][j] = __builtin_amdgcn_wmma_f32_16x16x32_bf16(
            false, a[i].v, false, b[j].v, (short)0, acc[i][j], false, false);
  }
#pragma unroll
  for (int i = 0; i < 2; ++i)
#pragma unroll
    for (int j = 0; j < 4; ++j)
#pragma unroll
      for (int v = 0; v < 8; ++v) {
        const int row = wave * 32 + i * 16 + v + 8 * hi;
        const int col = n0 + j * 16 + ln16;
        float val = acc[i][j][v];
        if (act) val = 0.5f * val * (1.0f + erff(val * 0.70710678118654752f));
        if (resmode == 1) val += res[(size_t)(row & 15) * Nf + col];
        else if (resmode == 2) val += res[(size_t)row * Nf + col];
        if (outf) outf[(size_t)row * Nf + col] = val;
        else      outb[(size_t)row * Nf + col] = (__bf16)val;
      }
}

// ---------------------------------------------------------------------------
// q = ql(16x4096 fp32) @ wq(4096x768) -> fp32 (computed once; latents invariant
// over b,t). Grid = 12 blocks x 64 cols; VALU fp32 keeps q full precision.
// ---------------------------------------------------------------------------
__global__ __launch_bounds__(256) void gemm_q_k(
    const float* __restrict__ ql, const float* __restrict__ wq,
    float* __restrict__ q) {
  __shared__ float qs[16 * 128];
  const int t = threadIdx.x;
  const int col = blockIdx.x * 64 + (t & 63);
  const int ig = t >> 6;  // 0..3 -> rows 4*ig..4*ig+3
  float acc[4] = {0.f, 0.f, 0.f, 0.f};
  for (int k0 = 0; k0 < DD; k0 += 128) {
    __syncthreads();
#pragma unroll
    for (int u = 0; u < 8; ++u) {
      const int c = t + 256 * u;            // 2048 = 16*128
      const int r = c >> 7, k = c & 127;
      qs[r * 128 + k] = ql[(size_t)r * DD + k0 + k];
    }
    __syncthreads();
    for (int k = 0; k < 128; ++k) {
      const float wv = wq[(size_t)(k0 + k) * INNERC + col];
#pragma unroll
      for (int r = 0; r < 4; ++r) acc[r] += wv * qs[(ig * 4 + r) * 128 + k];
    }
  }
#pragma unroll
  for (int r = 0; r < 4; ++r)
    q[(size_t)(ig * 4 + r) * INNERC + col] = acc[r];
}

// ---------------------------------------------------------------------------
// Attention for one (h, bt): sim = (q*s)K^T, max-shift, mask -> -FLT_MAX,
// * guidance, softmax, @V. K/V read bf16 from L2-resident kv buffer.
// ---------------------------------------------------------------------------
__global__ __launch_bounds__(256) void attn_k(
    const float* __restrict__ q, const __bf16* __restrict__ kv,
    const float* __restrict__ guid, const unsigned char* __restrict__ mask,
    __bf16* __restrict__ out) {
  const int h = blockIdx.x, bt = blockIdx.y;
  const int t = threadIdx.x;
  __shared__ float qs[16 * 96];
  __shared__ float ps[16 * 32];
  __shared__ float vs[32 * 96];
  __shared__ float red[256];
  __shared__ float smax[16];
  __shared__ float sden[16];
#pragma unroll
  for (int u = 0; u < 6; ++u) {
    const int c = t + 256 * u;  // exactly 1536 = 16*96
    qs[c] = q[(size_t)(c / 96) * INNERC + h * DHD + (c % 96)] * QSCALE;
  }
  __syncthreads();
  // ---- pass 1: row maxima of raw sim (pre-mask, matching reference) ----
  float mloc[16];
#pragma unroll
  for (int i = 0; i < 16; ++i) mloc[i] = -3.4028235e38f;
  for (int j0 = 0; j0 < NN; j0 += 256) {
    const int j = j0 + t;
    const __bf16* kp = kv + ((size_t)(bt * NN + j)) * KVC + h * DHD;
    float s[16];
#pragma unroll
    for (int i = 0; i < 16; ++i) s[i] = 0.f;
    for (int d = 0; d < DHD; ++d) {
      const float kvv = (float)kp[d];
#pragma unroll
      for (int i = 0; i < 16; ++i) s[i] += kvv * qs[i * 96 + d];
    }
#pragma unroll
    for (int i = 0; i < 16; ++i) mloc[i] = fmaxf(mloc[i], s[i]);
  }
  for (int i = 0; i < 16; ++i) {
    red[t] = mloc[i];
    __syncthreads();
    for (int off = 128; off > 0; off >>= 1) {
      if (t < off) red[t] = fmaxf(red[t], red[t + off]);
      __syncthreads();
    }
    if (t == 0) smax[i] = red[0];
    __syncthreads();
  }
  // ---- pass 2: p = exp(((sim-max) or -MAX)*g), denom, P@V ----
  const int i_out = t >> 4, dg = t & 15;
  const int jj = t & 31, ibase = (t >> 5) * 2;
  float o[6] = {0.f, 0.f, 0.f, 0.f, 0.f, 0.f};
  float den = 0.f;
  for (int j0 = 0; j0 < NN; j0 += 32) {
    __syncthreads();
    {
      const int j = j0 + jj;
      const __bf16* kp = kv + ((size_t)(bt * NN + j)) * KVC + h * DHD;
      float s0 = 0.f, s1 = 0.f;
      for (int d = 0; d < DHD; ++d) {
        const float kvv = (float)kp[d];
        s0 += kvv * qs[ibase * 96 + d];
        s1 += kvv * qs[(ibase + 1) * 96 + d];
      }
      const float g = guid[(size_t)bt * NN + j];
      const bool m = mask[(size_t)bt * NN + j] != 0;
      const float a0 = m ? -3.4028235e38f : (s0 - smax[ibase]);
      const float a1 = m ? -3.4028235e38f : (s1 - smax[ibase + 1]);
      ps[ibase * 32 + jj]       = __expf(a0 * g);
      ps[(ibase + 1) * 32 + jj] = __expf(a1 * g);
    }
#pragma unroll
    for (int u = 0; u < 12; ++u) {
      const int c = t + 256 * u;  // 3072 = 32*96
      const int vj = c / 96, vd = c % 96;
      vs[c] = (float)kv[((size_t)(bt * NN + j0 + vj)) * KVC + INNERC + h * DHD + vd];
    }
    __syncthreads();
#pragma unroll 4
    for (int jj2 = 0; jj2 < 32; ++jj2) {
      const float p = ps[i_out * 32 + jj2];
      if (dg == 0) den += p;
#pragma unroll
      for (int u = 0; u < 6; ++u) o[u] += p * vs[jj2 * 96 + dg * 6 + u];
    }
  }
  if (dg == 0) sden[i_out] = den;
  __syncthreads();
  const float inv = 1.0f / sden[i_out];
#pragma unroll
  for (int u = 0; u < 6; ++u)
    out[(size_t)(bt * 16 + i_out) * INNERC + h * DHD + dg * 6 + u] =
        (__bf16)(o[u] * inv);
}

// ---------------------------------------------------------------------------
extern "C" void kernel_launch(void* const* d_in, const int* in_sizes, int n_in,
                              void* d_out, int out_size, void* d_ws, size_t ws_size,
                              hipStream_t stream) {
  const float* x       = (const float*)d_in[0];
  const float* guid    = (const float*)d_in[1];
  const unsigned char* msk = (const unsigned char*)d_in[2];
  const float* latents = (const float*)d_in[3];
  const float* nm_g = (const float*)d_in[4];
  const float* nm_b = (const float*)d_in[5];
  const float* nl_g = (const float*)d_in[6];
  const float* nl_b = (const float*)d_in[7];
  const float* wq   = (const float*)d_in[8];
  const float* wkv  = (const float*)d_in[9];
  const float* wout = (const float*)d_in[10];
  const float* ff_g = (const float*)d_in[11];
  const float* ff_b = (const float*)d_in[12];
  const float* w1   = (const float*)d_in[13];
  const float* w2   = (const float*)d_in[14];
  const float* fn_g = (const float*)d_in[15];
  const float* fn_b = (const float*)d_in[16];

  char* ws = (char*)d_ws;
  size_t off = 0;
  __bf16* xm   = (__bf16*)(ws + off); off += (size_t)ROWSC * DD * 2;   // 235 MB
  __bf16* wkvT = (__bf16*)(ws + off); off += (size_t)KVC * DD * 2;     // 12.6 MB
  __bf16* kvb  = (__bf16*)(ws + off); off += (size_t)ROWSC * KVC * 2;  // 88 MB
  float*  ql   = (float*) (ws + off); off += (size_t)16 * DD * 4;
  float*  qq   = (float*) (ws + off); off += (size_t)16 * INNERC * 4;
  __bf16* aout = (__bf16*)(ws + off); off += (size_t)LROWS * INNERC * 2;
  float*  lat1 = (float*) (ws + off); off += (size_t)LROWS * DD * 4;
  __bf16* a2   = (__bf16*)(ws + off); off += (size_t)LROWS * DD * 2;
  __bf16* h1   = (__bf16*)(ws + off); off += (size_t)LROWS * FFD * 2;
  float*  lat2 = (float*) (ws + off); off += (size_t)LROWS * DD * 4;

  // 1) LN(x) -> bf16
  ln_rows_k<<<ROWSC, 256, 0, stream>>>(x, nm_g, nm_b, nullptr, xm);
  // 2) wkv -> bf16 transposed
  cvt_wkvT_k<<<(KVC * DD) / 256, 256, 0, stream>>>(wkv, wkvT);
  // 3) LN(latents) -> fp32 (shared across all b,t)
  ln_rows_k<<<16, 256, 0, stream>>>(latents, nl_g, nl_b, ql, nullptr);
  // 4) q = ql @ wq (once)
  gemm_q_k<<<INNERC / 64, 256, 0, stream>>>(ql, wq, qq);
  // 5) kv = xm @ wkv   (WMMA bf16 + double-buffered async-to-LDS)
  gemm_kv_k<<<dim3(KVC / 64, ROWSC / 128), 256, 0, stream>>>(xm, wkvT, kvb);
  // 6) attention per (h, bt); kv is L2-resident
  attn_k<<<dim3(HH, BTC), 256, 0, stream>>>(qq, kvb, guid, msk, aout);
  // 7) lat1 = aout @ wout + latents(broadcast)
  gemm_wb_k<<<DD / 64, 256, 0, stream>>>(aout, INNERC, wout, DD,
                                         latents, 1, 0, lat1, nullptr);
  // 8) LN(lat1) -> bf16
  ln_rows_k<<<LROWS, 256, 0, stream>>>(lat1, ff_g, ff_b, nullptr, a2);
  // 9) h1 = gelu(a2 @ w1)  (bf16 out)
  gemm_wb_k<<<FFD / 64, 256, 0, stream>>>(a2, DD, w1, FFD,
                                          nullptr, 0, 1, nullptr, h1);
  // 10) lat2 = h1 @ w2 + lat1
  gemm_wb_k<<<DD / 64, 256, 0, stream>>>(h1, FFD, w2, DD,
                                         lat1, 2, 0, lat2, nullptr);
  // 11) final LN -> d_out (fp32)
  ln_rows_k<<<LROWS, 256, 0, stream>>>(lat2, fn_g, fn_b, (float*)d_out, nullptr);
}
